// MeshUnpool_77438260347528
// MI455X (gfx1250) — matrix operand order repaired and checked by reading, not compiled, not run on hardware
//
#include <hip/hip_runtime.h>
#include <hip/hip_bf16.h>

typedef _Float16 half_t;
typedef __attribute__((ext_vector_type(16))) _Float16 v16h;
typedef __attribute__((ext_vector_type(8)))  _Float16 v8h;
typedef __attribute__((ext_vector_type(8)))  float    v8f;

#define BATCH 8
#define NF    128
#define EDGES 3072
#define UDIM  4096
#define NT    64     // output columns per block
#define KT    32     // K (v) per step == WMMA f16 K
#define ROWH  40     // 32 K-halves + 8 pad; row stride = 80 B (16B aligned)

// -------- Kernel 1: per-batch inclusive scan of dst mask -> gather indices ----
__global__ void scan_idx_kernel(const unsigned char* __restrict__ dst,
                                int* __restrict__ idx) {
  __shared__ int s[1024];
  const int b = blockIdx.x;
  const int t = threadIdx.x;                 // 1024 threads, 4 elems each
  const unsigned char* d = dst + (size_t)b * UDIM;
  const int v0 = t * 4;
  const int c0 = d[v0 + 0] ? 1 : 0;
  const int c1 = d[v0 + 1] ? 1 : 0;
  const int c2 = d[v0 + 2] ? 1 : 0;
  const int c3 = d[v0 + 3] ? 1 : 0;
  const int loc = c0 + c1 + c2 + c3;
  s[t] = loc;
  __syncthreads();
  for (int off = 1; off < 1024; off <<= 1) {   // Hillis-Steele inclusive scan
    int val = (t >= off) ? s[t - off] : 0;
    __syncthreads();
    s[t] += val;
    __syncthreads();
  }
  int run = s[t] - loc;                        // exclusive prefix
  int* o = idx + (size_t)b * UDIM + v0;
  run += c0; o[0] = min(max(run - 1, 0), EDGES - 1);
  run += c1; o[1] = min(max(run - 1, 0), EDGES - 1);
  run += c2; o[2] = min(max(run - 1, 0), EDGES - 1);
  run += c3; o[3] = min(max(run - 1, 0), EDGES - 1);
}

// -------- Kernel 2: build gathered/masked A as hi+lo f16 planes --------------
// A[b][n][v] = dst ? features[b][n][idx[v]] : 0 ; a = hi + lo (exact split)
__global__ void build_a_kernel(const float* __restrict__ feats,
                               const unsigned char* __restrict__ dst,
                               const int* __restrict__ idx,
                               half_t* __restrict__ Ahi,
                               half_t* __restrict__ Alo) {
  const size_t total = (size_t)BATCH * NF * UDIM;
  for (size_t i = (size_t)blockIdx.x * blockDim.x + threadIdx.x; i < total;
       i += (size_t)gridDim.x * blockDim.x) {
    const int    v  = (int)(i & (UDIM - 1));
    const size_t bn = i >> 12;                 // b*NF + n  (UDIM = 2^12)
    const int    b  = (int)(bn >> 7);          // NF = 2^7
    const size_t bv = (size_t)b * UDIM + v;
    float val = 0.0f;
    if (dst[bv]) val = feats[bn * EDGES + (size_t)idx[bv]];
    const half_t hi = (half_t)val;
    const half_t lo = (half_t)(val - (float)hi);
    Ahi[i] = hi;
    Alo[i] = lo;
  }
}

__device__ __forceinline__ v16h cat16(v8h a, v8h b) {
  v16h r;
#pragma unroll
  for (int i = 0; i < 8; ++i) { r[i] = a[i]; r[i + 8] = b[i]; }
  return r;
}

// -------- Kernel 3: batched WMMA GEMM: out = (Ahi+Alo) x unroll, / occ -------
__global__ __launch_bounds__(256)
void unpool_gemm_kernel(const float* __restrict__ unroll,
                        const float* __restrict__ occ,
                        const half_t* __restrict__ Ahi,
                        const half_t* __restrict__ Alo,
                        float* __restrict__ out) {
  __shared__ __attribute__((aligned(16))) half_t ldsB[2][NT * ROWH];

  const int b     = blockIdx.y;
  const int uBase = blockIdx.x * NT;
  const int t     = threadIdx.x;        // 256 threads = 8 waves
  const int wave  = t >> 5;
  const int lane  = t & 31;
  const int g     = wave >> 1;          // M group: rows 32g .. 32g+31
  const int h     = wave & 1;           // N group: cols 32h .. 32h+31
  const int lrow  = lane & 15;
  const int lhalf = (lane < 16) ? 0 : 1;

  const float* Ub = unroll + (size_t)b * UDIM * UDIM + uBase;  // + k*U + u

  // B-tile staging assignment: each thread owns column u=t&63, 8 contiguous k
  const int ucol_ld = t & 63;
  const int krow    = (t >> 6) << 3;    // 0,8,16,24

  v8f acc[2][2] = {};
  float regs[8];

  // Prologue: load + stage tile 0
  {
    const float* src = Ub;
#pragma unroll
    for (int i = 0; i < 8; ++i) regs[i] = src[(size_t)(krow + i) * UDIM + ucol_ld];
    v8h hv;
#pragma unroll
    for (int i = 0; i < 8; ++i) hv[i] = (half_t)regs[i];
    *(v8h*)&ldsB[0][ucol_ld * ROWH + krow] = hv;
  }
  __syncthreads();

  const half_t* AhiB = Ahi + (size_t)b * NF * UDIM;
  const half_t* AloB = Alo + (size_t)b * NF * UDIM;

  int buf = 0;
  const int KTILES = UDIM / KT;         // 128
  for (int kk = 0; kk < KTILES; ++kk) {
    const bool more = (kk + 1 < KTILES);
    if (more) {                         // issue next tile's global loads early
      const float* src = Ub + (size_t)((kk + 1) * KT) * UDIM;
#pragma unroll
      for (int i = 0; i < 8; ++i) regs[i] = src[(size_t)(krow + i) * UDIM + ucol_ld];
    }
    if (kk + 2 < KTILES) {              // prefetch tile after next into GL2
      const float* pf = Ub + (size_t)((kk + 2) * KT) * UDIM;
      __builtin_prefetch(pf + (size_t)(t >> 3) * UDIM + (t & 7) * 8, 0, 1);
    }

    const int kBase = kk * KT;
    // A fragments (hi/lo): 16-bit A 16x32 layout -> per lane two 16B chunks:
    // K = koff..koff+7 and koff+16..koff+23, koff = (lane<16 ? 0 : 8)
    v16h ah[2], al[2], bf[2];
#pragma unroll
    for (int mt = 0; mt < 2; ++mt) {
      const int m = g * 32 + mt * 16 + lrow;
      const half_t* ph = AhiB + (size_t)m * UDIM + kBase + lhalf * 8;
      const half_t* pl = AloB + (size_t)m * UDIM + kBase + lhalf * 8;
      ah[mt] = cat16(*(const v8h*)ph, *(const v8h*)(ph + 16));
      al[mt] = cat16(*(const v8h*)pl, *(const v8h*)(pl + 16));
    }
    // B fragments from transposed LDS tile: per lane K = k0..k0+7 and k0+16..k0+23,
    // k0 = (lane<16 ? 0 : 8), column n = tileBase + (lane&15)
    #pragma unroll
    for (int nt = 0; nt < 2; ++nt) {
      const int uc = h * 32 + nt * 16 + lrow;
      const half_t* pb = &ldsB[buf][uc * ROWH + lhalf * 8];
      bf[nt] = cat16(*(const v8h*)pb, *(const v8h*)(pb + 16));
    }
#pragma unroll
    for (int mt = 0; mt < 2; ++mt)
#pragma unroll
      for (int nt = 0; nt < 2; ++nt) {
        acc[mt][nt] = __builtin_amdgcn_wmma_f32_16x16x32_f16(
            false, ah[mt], false, bf[nt], (short)0, acc[mt][nt], false, false);
        acc[mt][nt] = __builtin_amdgcn_wmma_f32_16x16x32_f16(
            false, al[mt], false, bf[nt], (short)0, acc[mt][nt], false, false);
      }

    if (more) {                         // stage next tile into other buffer
      v8h hv;
#pragma unroll
      for (int i = 0; i < 8; ++i) hv[i] = (half_t)regs[i];
      *(v8h*)&ldsB[buf ^ 1][ucol_ld * ROWH + krow] = hv;
    }
    __syncthreads();
    buf ^= 1;
  }

  // Epilogue: divide by occurrences, store f32.
  // C/D layout: VGPR i -> M = i (lanes 0-15) or i+8 (lanes 16-31); N = lane&15
  const float* occB = occ + (size_t)b * UDIM;
  float* outB = out + (size_t)b * NF * UDIM;
#pragma unroll
  for (int nt = 0; nt < 2; ++nt) {
    const int uc = uBase + h * 32 + nt * 16 + lrow;
    const float ov = occB[uc];
#pragma unroll
    for (int mt = 0; mt < 2; ++mt) {
#pragma unroll
      for (int i = 0; i < 8; ++i) {
        const int m = g * 32 + mt * 16 + (lhalf ? i + 8 : i);
        outB[(size_t)m * UDIM + uc] = acc[mt][nt][i] / ov;
      }
    }
  }
}

extern "C" void kernel_launch(void* const* d_in, const int* in_sizes, int n_in,
                              void* d_out, int out_size, void* d_ws, size_t ws_size,
                              hipStream_t stream) {
  const float*         feats  = (const float*)d_in[0];         // [B,NF,E]
  const float*         unroll = (const float*)d_in[1];         // [B,U,U]
  const float*         occ    = (const float*)d_in[2];         // [B,1,U]
  const unsigned char* dst    = (const unsigned char*)d_in[3]; // [B,U] bool
  float* out = (float*)d_out;                                  // [B,NF,U]

  const size_t planeElems = (size_t)BATCH * NF * UDIM;         // 4.19M halves
  half_t* Ahi = (half_t*)d_ws;                                 // 8 MB
  half_t* Alo = Ahi + planeElems;                              // 8 MB
  int*    idx = (int*)(Alo + planeElems);                      // 128 KB

  scan_idx_kernel<<<BATCH, 1024, 0, stream>>>(dst, idx);
  build_a_kernel<<<4096, 256, 0, stream>>>(feats, dst, idx, Ahi, Alo);

  dim3 grid(UDIM / NT, BATCH);                                 // 64 x 8 blocks
  unpool_gemm_kernel<<<grid, 256, 0, stream>>>(unroll, occ, Ahi, Alo, out);
}
